// UA_Aggregator_90829968376432
// MI455X (gfx1250) — matrix operand order repaired and checked by reading, not compiled, not run on hardware
//
#include <hip/hip_runtime.h>
#include <hip/hip_bf16.h>

typedef __attribute__((ext_vector_type(16))) __bf16 v16bf;
typedef __attribute__((ext_vector_type(8)))  __bf16 v8bf;
typedef __attribute__((ext_vector_type(8)))  float  v8f;

#define B_N   2048
#define L_N   200
#define LPAD  208     // 13 m-tiles of 16
#define MT_N  13
#define D_N   256
#define D2_N  512

// ---------------------------------------------------------------------------
// Kernel 0: one-shot f32 -> bf16 weight conversion into workspace.
// Layout in ws (elements): [0,131072) W1 | [131072,196608) W2
//                          [196608,327680) Att1 | [327680,393216) Att2
// ---------------------------------------------------------------------------
__global__ __launch_bounds__(256)
void ua_convert_weights(const float* __restrict__ w1, const float* __restrict__ w2,
                        const float* __restrict__ a1, const float* __restrict__ a2,
                        __bf16* __restrict__ out) {
    int i = blockIdx.x * 256 + threadIdx.x;   // grid covers exactly 393216
    float v;
    if (i < 131072)       v = w1[i];
    else if (i < 196608)  v = w2[i - 131072];
    else if (i < 327680)  v = a1[i - 196608];
    else                  v = a2[i - 327680];
    out[i] = (__bf16)v;
}

// ---------------------------------------------------------------------------
// WMMA helpers (CDNA5 V_WMMA_F32_16X16X32_BF16 lane layouts)
// ---------------------------------------------------------------------------
static __device__ inline v16bf cat8(v8bf lo, v8bf hi) {
    return __builtin_shufflevector(lo, hi, 0,1,2,3,4,5,6,7,8,9,10,11,12,13,14,15);
}

// A: 16 x 32 bf16 tile from a row-major [16 x kdim] LDS stage.
// lane m (<16): K k0..k0+15; lane m+16: K k0+16..k0+31
static __device__ inline v16bf load_a(const __bf16* p /* row base + half*16 + k0 */) {
    v8bf lo = *reinterpret_cast<const v8bf*>(p);
    v8bf hi = *reinterpret_cast<const v8bf*>(p + 8);
    return cat8(lo, hi);
}

// B: 32 x 16 tile with B[k,n] = W[n,k].
// lane n (<16) holds K k0..k0+15 of W row n; lane n+16 holds K k0+16..k0+31.
// => contiguous 32-byte chunk of the bf16 weight row (coalesced from L2).
static __device__ inline v16bf load_b(const __bf16* wrow_chunk) {
    v8bf lo = *reinterpret_cast<const v8bf*>(wrow_chunk);
    v8bf hi = *reinterpret_cast<const v8bf*>(wrow_chunk + 8);
    return cat8(lo, hi);
}

// One wave computes two 16x16 output tiles (nt0, nt0+1) of  IN[16 x kdim] * W^T.
// The *offsets* are laundered through an empty asm: this blocks LICM from
// hoisting all weight tiles out of the m-tile loop (round-1 scratch-spill
// disaster) while keeping the base pointer's global address space visible,
// so the loads lower to global_load_b128 (saddr+voffset), not flat_load.
static __device__ inline void gemm2(const __bf16* __restrict__ Astage,
                                    const __bf16* __restrict__ Wb, int kdim,
                                    int nt0, int row, int half,
                                    v8f& acc0, v8f& acc1) {
    int off0 = (nt0 * 16 + row) * kdim + half * 16;
    int off1 = off0 + 16 * kdim;
    asm volatile("" : "+v"(off0), "+v"(off1));
    const __bf16* a0 = Astage + row * kdim + half * 16;
    #pragma unroll 2
    for (int k0 = 0; k0 < kdim; k0 += 32) {
        v16bf a  = load_a(a0 + k0);
        v16bf b0 = load_b(Wb + off0 + k0);
        v16bf b1 = load_b(Wb + off1 + k0);
        acc0 = __builtin_amdgcn_wmma_f32_16x16x32_bf16(false, a, false, b0,
                                                       (short)0, acc0, false, false);
        acc1 = __builtin_amdgcn_wmma_f32_16x16x32_bf16(false, a, false, b1,
                                                       (short)0, acc1, false, false);
    }
}

// C/D layout: VGPR i -> row (i + 8*half), col nt*16 + (lane&15)
static __device__ inline void store_tile(const v8f& acc, const float* __restrict__ bias,
                                         int nt, int row, int half,
                                         __bf16* dst, int ld, bool relu) {
    int n = nt * 16 + row;
    float bb = bias[n];
    #pragma unroll
    for (int i = 0; i < 8; ++i) {
        int m = i + half * 8;
        float v = acc[i] + bb;
        if (relu) v = fmaxf(v, 0.0f);
        dst[m * ld + n] = (__bf16)v;
    }
}

// ---------------------------------------------------------------------------
// Main fused kernel: one workgroup per batch element.
// ---------------------------------------------------------------------------
__global__ __launch_bounds__(256)
void ua_aggregator_kernel(const int*   __restrict__ nodes,
                          const int*   __restrict__ hist_ua,
                          const int*   __restrict__ hist_r,
                          const float* __restrict__ u2e,
                          const float* __restrict__ attr,
                          const float* __restrict__ r2e,
                          const __bf16* __restrict__ Wcvt,
                          const float* __restrict__ b1,
                          const float* __restrict__ b2,
                          const float* __restrict__ ba1,
                          const float* __restrict__ ba2,
                          const float* __restrict__ att3w,
                          const float* __restrict__ att3b,
                          float* __restrict__ out) {
    __shared__ __align__(16) __bf16 Obuf[LPAD * D_N];   // 106496 B : o history (bf16)
    __shared__ __align__(16) __bf16 Xs[16 * D2_N];      //  16384 B : gather stage / H3 stage
    __shared__ __align__(16) __bf16 As[16 * D2_N];      //  16384 B : [o | ua_rep] stage
    __shared__ __align__(16) __bf16 Hs[16 * D_N];       //   8192 B : inter-layer stage
    __shared__ float scores[LPAD];
    __shared__ int   idxa[16], idxr[16];

    const __bf16* W1b = Wcvt;
    const __bf16* W2b = Wcvt + 131072;
    const __bf16* A1b = Wcvt + 196608;
    const __bf16* A2b = Wcvt + 327680;

    const int b    = blockIdx.x;
    const int tid  = threadIdx.x;
    const int lane = tid & 31;
    const int wv   = tid >> 5;       // 8 waves
    const int row  = lane & 15;
    const int half = lane >> 4;
    const int nt0  = wv * 2;         // each wave owns 2 of 16 N-tiles

    // ua_rep -> second half of attention input (constant across all m-tiles)
    {
        int node = nodes[b];
        for (int e = tid; e < 16 * D_N; e += 256) {
            int m = e >> 8, c = e & 255;
            As[m * D2_N + D_N + c] = (__bf16)u2e[node * D_N + c];
        }
    }

    for (int mt = 0; mt < MT_N; ++mt) {
        __syncthreads();
        if (tid < 16) {
            int l  = mt * 16 + tid;
            int lc = l < L_N ? l : L_N - 1;
            idxa[tid] = hist_ua[b * L_N + lc];
            idxr[tid] = hist_r[b * L_N + lc];
        }
        __syncthreads();

        // gather + f32->bf16: Xs[m] = [attr_w[ua] | r2e_w[r]]
        for (int e = tid; e < 16 * D2_N; e += 256) {
            int m = e >> 9, c = e & 511;
            float v = (c < D_N) ? attr[idxa[m] * D_N + c]
                                : r2e[idxr[m] * D_N + (c - D_N)];
            Xs[e] = (__bf16)v;
        }
        __syncthreads();

        // layer 1: Hs = relu(Xs * W1^T + b1)       [16x512]x[512x256]
        {
            v8f acc0 = {}; v8f acc1 = {};
            gemm2(Xs, W1b, D2_N, nt0, row, half, acc0, acc1);
            store_tile(acc0, b1, nt0,     row, half, Hs, D_N, true);
            store_tile(acc1, b1, nt0 + 1, row, half, Hs, D_N, true);
        }
        __syncthreads();

        // layer 2: o = relu(Hs * W2^T + b2)        [16x256]x[256x256]
        {
            v8f acc0 = {}; v8f acc1 = {};
            gemm2(Hs, W2b, D_N, nt0, row, half, acc0, acc1);
            store_tile(acc0, b2, nt0,     row, half, Obuf + mt * 16 * D_N, D_N, true);
            store_tile(acc1, b2, nt0 + 1, row, half, Obuf + mt * 16 * D_N, D_N, true);
            store_tile(acc0, b2, nt0,     row, half, As, D2_N, true);   // first half of att input
            store_tile(acc1, b2, nt0 + 1, row, half, As, D2_N, true);
        }
        __syncthreads();

        // att layer 1: Hs = relu(As * Att1^T + ba1) [16x512]x[512x256]
        {
            v8f acc0 = {}; v8f acc1 = {};
            gemm2(As, A1b, D2_N, nt0, row, half, acc0, acc1);
            store_tile(acc0, ba1, nt0,     row, half, Hs, D_N, true);
            store_tile(acc1, ba1, nt0 + 1, row, half, Hs, D_N, true);
        }
        __syncthreads();

        // att layer 2: H3 = relu(Hs * Att2^T + ba2) -> Xs (reused, ld=256)
        {
            v8f acc0 = {}; v8f acc1 = {};
            gemm2(Hs, A2b, D_N, nt0, row, half, acc0, acc1);
            store_tile(acc0, ba2, nt0,     row, half, Xs, D_N, true);
            store_tile(acc1, ba2, nt0 + 1, row, half, Xs, D_N, true);
        }
        __syncthreads();

        // score[l] = H3[l] . att3 + b
        if (tid < 16) {
            int l = mt * 16 + tid;
            float s = att3b[0];
            for (int j = 0; j < D_N; ++j)
                s += (float)Xs[tid * D_N + j] * att3w[j];
            scores[l] = (l < L_N) ? s : -1e30f;
        }
    }
    __syncthreads();

    // softmax over L (redundant-uniform max/sum, then per-thread exp)
    float mx = -1e30f;
    for (int l = 0; l < LPAD; ++l) mx = fmaxf(mx, scores[l]);
    float sum = 0.0f;
    for (int l = 0; l < LPAD; ++l) sum += __expf(scores[l] - mx);
    __syncthreads();
    if (tid < LPAD) scores[tid] = __expf(scores[tid] - mx);
    __syncthreads();

    // out[b, d] = (1/sum) * sum_l exp(s_l - mx) * o[l, d]
    float acc = 0.0f;
    for (int l = 0; l < L_N; ++l)
        acc += scores[l] * (float)Obuf[l * D_N + tid];
    out[b * D_N + tid] = acc / sum;
}

// ---------------------------------------------------------------------------
extern "C" void kernel_launch(void* const* d_in, const int* in_sizes, int n_in,
                              void* d_out, int out_size, void* d_ws, size_t ws_size,
                              hipStream_t stream) {
    (void)in_sizes; (void)n_in; (void)out_size; (void)ws_size;
    const int*   nodes = (const int*)  d_in[0];
    const int*   hua   = (const int*)  d_in[1];
    const int*   hr    = (const int*)  d_in[2];
    /* d_in[3] history_uat unused by reference */
    const float* u2e   = (const float*)d_in[4];
    const float* attr  = (const float*)d_in[5];
    const float* r2e   = (const float*)d_in[6];
    const float* w1    = (const float*)d_in[7];
    const float* b1    = (const float*)d_in[8];
    const float* w2    = (const float*)d_in[9];
    const float* b2    = (const float*)d_in[10];
    const float* a1    = (const float*)d_in[11];
    const float* ba1   = (const float*)d_in[12];
    const float* a2    = (const float*)d_in[13];
    const float* ba2   = (const float*)d_in[14];
    const float* a3    = (const float*)d_in[15];
    const float* ba3   = (const float*)d_in[16];

    __bf16* wb = (__bf16*)d_ws;   // 786432 bytes of bf16 weights

    ua_convert_weights<<<393216 / 256, 256, 0, stream>>>(w1, w2, a1, a2, wb);
    ua_aggregator_kernel<<<B_N, 256, 0, stream>>>(nodes, hua, hr, u2e, attr, r2e, wb,
                                                  b1, b2, ba1, ba2, a3, ba3,
                                                  (float*)d_out);
}
